// GRU_14345190768816
// MI455X (gfx1250) — compile-verified, matrix operand order checked by reading, not run on hardware
//
#include <hip/hip_runtime.h>
#include <hip/hip_bf16.h>
#include <math.h>

// ---- problem constants (match reference) ----
#define B_  256
#define S_  256
#define D_  512
#define H_  1024
#define KT_ 1536   // D + H

#define KSLAB   128                 // K elements staged per TDM transfer
#define NSLAB   (KT_ / KSLAB)       // 12 slabs; slabs [0,4) are x, [4,12) are h
#define NXSLAB  (D_ / KSLAB)        // 4
#define COLSLOT 320                 // bytes per column in LDS: 4 x (64B data + 16B pad)
#define SLABBYTES (64 * COLSLOT)    // 20480 B per slab buffer

typedef __attribute__((ext_vector_type(16))) __bf16 v16bf;
typedef __attribute__((ext_vector_type(8)))  float  v8f;
typedef __attribute__((ext_vector_type(4)))  float  f32x4;
typedef __attribute__((ext_vector_type(4)))  unsigned int u32x4_t;
typedef __attribute__((ext_vector_type(4)))  int    i32x4_t;
typedef __attribute__((ext_vector_type(8)))  int    i32x8_t;

// ---- TDM availability (device pass only; host pass parses the fallback) ----
#if defined(__HIP_DEVICE_COMPILE__) && defined(__gfx1250__)
  #if __has_builtin(__builtin_amdgcn_tensor_load_to_lds) && __has_builtin(__builtin_amdgcn_s_wait_tensorcnt)
    #define USE_TDM 1
  #endif
#endif
#ifndef USE_TDM
  #define USE_TDM 0
#endif
#if __has_include(<hip/amd_detail/amd_gfx1250_TDM.h>)
  #define TDM_ARITY6 1
#else
  #define TDM_ARITY6 0
#endif

__device__ __forceinline__ float sigmoidf_(float v) {
    return 1.0f / (1.0f + __expf(-v));
}

// Load 16 contiguous f32, convert to 16 bf16 (x-sourced A-fragment payload).
__device__ __forceinline__ v16bf cvt16(const float* __restrict__ p) {
    f32x4 f[4];
#pragma unroll
    for (int q = 0; q < 4; ++q) f[q] = *(const f32x4*)(p + 4 * q);
    v16bf a;
#pragma unroll
    for (int q = 0; q < 4; ++q) {
        a[4 * q + 0] = (__bf16)f[q].x;
        a[4 * q + 1] = (__bf16)f[q].y;
        a[4 * q + 2] = (__bf16)f[q].z;
        a[4 * q + 3] = (__bf16)f[q].w;
    }
    return a;
}

#if USE_TDM
// Issue one TDM transfer: 64 columns x 128 K of bf16 weights into LDS,
// padded so each 32-K group sits in an 80B (16B-aligned, bank-rotated) slot.
__device__ __forceinline__ void tdm_issue(void* ldsdst, const __bf16* gsrc) {
    u32x4_t g0;
    unsigned long long ga = (unsigned long long)(size_t)gsrc;
    g0[0] = 1u;                                   // count=1, user descriptor
    g0[1] = (unsigned)(size_t)ldsdst;             // lds_addr (low 32 bits = LDS offset)
    g0[2] = (unsigned)ga;                         // global_addr[31:0]
    g0[3] = (unsigned)((ga >> 32) & 0x01FFFFFFu)  // global_addr[56:32]
          | 0x80000000u;                          // type = 2 (image)
    i32x8_t g1;
    g1[0] = (int)((1u << 16)                      // data_size = 2 bytes
          | (1u << 20)                            // pad_enable
          | (3u << 22)                            // pad_interval: 16 dwords (64B)
          | (3u << 25));                          // pad_amount:   4 dwords (16B)
    g1[1] = (int)(128u << 16);                    // tensor_dim0 = 128 (K)
    g1[2] = (int)(64u << 16);                     // tensor_dim1 = 64  (cols)
    g1[3] = (int)(128u << 16);                    // tile_dim0   = 128
    g1[4] = 64;                                   // tile_dim1   = 64, tile_dim2 = 0
    g1[5] = KT_;                                  // tensor_dim0_stride = 1536
    g1[6] = 0;
    g1[7] = 0;
    i32x4_t z4 = {};
#if TDM_ARITY6
    i32x8_t z8 = {};
    __builtin_amdgcn_tensor_load_to_lds(g0, g1, z4, z4, z8, 0);
#else
    __builtin_amdgcn_tensor_load_to_lds(g0, g1, z4, z4, 0);
#endif
}
#endif

// ---- one-time: transpose + convert the three weight matrices to bf16 [H][KT] ----
__global__ __launch_bounds__(256) void wconv_kernel(const float* __restrict__ Wr,
                                                    const float* __restrict__ Wz,
                                                    const float* __restrict__ Wt,
                                                    __bf16* __restrict__ out) {
    __shared__ float tile[16][17];
    const int which = blockIdx.z;
    const float* src = (which == 0) ? Wr : (which == 1) ? Wz : Wt;
    __bf16* dst = out + (size_t)which * H_ * KT_;
    const int k0 = blockIdx.x * 16, n0 = blockIdx.y * 16;
    const int tx = threadIdx.x, ty = threadIdx.y;
    tile[ty][tx] = src[(size_t)(k0 + ty) * H_ + (n0 + tx)];
    __syncthreads();
    dst[(size_t)(n0 + ty) * KT_ + (k0 + tx)] = (__bf16)tile[tx][ty];
}

__global__ __launch_bounds__(256) void zero_kernel(float* __restrict__ p,
                                                   __bf16* __restrict__ pb) {
    const int i = blockIdx.x * 256 + threadIdx.x;
    p[i] = 0.0f;
    pb[i] = (__bf16)0.0f;
}

// ---- per-step kernel 1: r,z gates. r-tiles additionally emit bf16 (r*h). ----
// grid (4, 32), block 128 (4 waves). Wave = 16(M) x 64(N). N tile uniform r or z.
__global__ __launch_bounds__(128) void rz_gemm_kernel(
    const float* __restrict__ x, int t,
    const float* __restrict__ hprev,        // fp32 h (for r*h product)
    const __bf16* __restrict__ hbf,         // bf16 shadow of h (A fragments)
    const __bf16* __restrict__ WrT, const __bf16* __restrict__ WzT,
    const float* __restrict__ br, const float* __restrict__ bz,
    __bf16* __restrict__ rhbf,              // out: bf16 (r*h)
    float* __restrict__ zbuf) {             // out: fp32 z
    __shared__ __align__(16) char lds[2][SLABBYTES];

    const int lane  = threadIdx.x & 31;
    const int wave  = threadIdx.x >> 5;
    const int m0    = blockIdx.x * 64 + wave * 16;   // batch rows
    const int n0    = blockIdx.y * 64;               // combined r|z column base
    const int arow  = m0 + (lane & 15);
    const int khalf = (lane >> 4) * 16;
    const int ncol  = lane & 15;

    const __bf16* wsel = (n0 < H_) ? (WrT + (size_t)n0 * KT_)
                                   : (WzT + (size_t)(n0 - H_) * KT_);
    v8f acc[4] = {};

#if USE_TDM
    if (threadIdx.x < 32) tdm_issue(&lds[0][0], wsel);
#endif
    for (int s = 0; s < NSLAB; ++s) {
        char* buf;
#if USE_TDM
        buf = &lds[s & 1][0];
        if (threadIdx.x < 32) {
            if (s + 1 < NSLAB) {
                tdm_issue(&lds[(s + 1) & 1][0], wsel + (size_t)(s + 1) * KSLAB);
                __builtin_amdgcn_s_wait_tensorcnt((short)1);
            } else {
                __builtin_amdgcn_s_wait_tensorcnt((short)0);
            }
        }
        __syncthreads();
#else
        buf = &lds[0][0];
        for (int u = threadIdx.x; u < 256; u += 128) {
            const int c = u >> 2, q = u & 3;
            const uint4* s4 = (const uint4*)(wsel + (size_t)c * KT_ +
                                             (size_t)s * KSLAB + q * 32);
            uint4* d4 = (uint4*)(buf + c * COLSLOT + q * 80);
            d4[0] = s4[0]; d4[1] = s4[1]; d4[2] = s4[2]; d4[3] = s4[3];
        }
        __syncthreads();
#endif
        if (s < NXSLAB) {
            const float* ab = x + ((size_t)arow * S_ + t) * D_ + (size_t)s * KSLAB;
#pragma unroll
            for (int q = 0; q < 4; ++q) {
                v16bf a = cvt16(ab + q * 32 + khalf);
                const char* bb = buf + q * 80 + khalf * 2;
#pragma unroll
                for (int j = 0; j < 4; ++j) {
                    v16bf b = *(const v16bf*)(bb + (16 * j + ncol) * COLSLOT);
                    acc[j] = __builtin_amdgcn_wmma_f32_16x16x32_bf16(
                        false, a, false, b, (short)0, acc[j], false, false);
                }
            }
        } else {
            const __bf16* ab = hbf + (size_t)arow * H_ + (size_t)(s - NXSLAB) * KSLAB;
#pragma unroll
            for (int q = 0; q < 4; ++q) {
                v16bf a = *(const v16bf*)(ab + q * 32 + khalf);
                const char* bb = buf + q * 80 + khalf * 2;
#pragma unroll
                for (int j = 0; j < 4; ++j) {
                    v16bf b = *(const v16bf*)(bb + (16 * j + ncol) * COLSLOT);
                    acc[j] = __builtin_amdgcn_wmma_f32_16x16x32_bf16(
                        false, a, false, b, (short)0, acc[j], false, false);
                }
            }
        }
        __syncthreads();
    }

    if (n0 < H_) {
        // r tiles: emit bf16 (r*h) directly for the candidate GEMM.
#pragma unroll
        for (int j = 0; j < 4; ++j) {
            const int nn = n0 + 16 * j + ncol;
            const float bv = br[nn];
#pragma unroll
            for (int i = 0; i < 8; ++i) {
                const int m = m0 + i + 8 * (lane >> 4);
                const float rv = sigmoidf_(acc[j][i] + bv);
                const float hv = hprev[(size_t)m * H_ + nn];
                rhbf[(size_t)m * H_ + nn] = (__bf16)(rv * hv);
            }
        }
    } else {
#pragma unroll
        for (int j = 0; j < 4; ++j) {
            const int nn = (n0 - H_) + 16 * j + ncol;
            const float bv = bz[nn];
#pragma unroll
            for (int i = 0; i < 8; ++i) {
                const int m = m0 + i + 8 * (lane >> 4);
                zbuf[(size_t)m * H_ + nn] = sigmoidf_(acc[j][i] + bv);
            }
        }
    }
}

// ---- per-step kernel 2: h_new = z*h + (1-z)*tanh([x_t, r*h] @ Wt + bt) ----
// grid (4, 16), block 128. Emits fp32 h_new plus its bf16 shadow.
__global__ __launch_bounds__(128) void ht_gemm_kernel(
    const float* __restrict__ x, int t,
    const float* __restrict__ hprev,
    const __bf16* __restrict__ rhbf,
    const float* __restrict__ zbuf,
    const __bf16* __restrict__ WtT, const float* __restrict__ bt,
    float* __restrict__ hnext, __bf16* __restrict__ hbfnext) {
    __shared__ __align__(16) char lds[2][SLABBYTES];

    const int lane  = threadIdx.x & 31;
    const int wave  = threadIdx.x >> 5;
    const int m0    = blockIdx.x * 64 + wave * 16;
    const int n0    = blockIdx.y * 64;
    const int arow  = m0 + (lane & 15);
    const int khalf = (lane >> 4) * 16;
    const int ncol  = lane & 15;

    const __bf16* wsel = WtT + (size_t)n0 * KT_;
    v8f acc[4] = {};

#if USE_TDM
    if (threadIdx.x < 32) tdm_issue(&lds[0][0], wsel);
#endif
    for (int s = 0; s < NSLAB; ++s) {
        char* buf;
#if USE_TDM
        buf = &lds[s & 1][0];
        if (threadIdx.x < 32) {
            if (s + 1 < NSLAB) {
                tdm_issue(&lds[(s + 1) & 1][0], wsel + (size_t)(s + 1) * KSLAB);
                __builtin_amdgcn_s_wait_tensorcnt((short)1);
            } else {
                __builtin_amdgcn_s_wait_tensorcnt((short)0);
            }
        }
        __syncthreads();
#else
        buf = &lds[0][0];
        for (int u = threadIdx.x; u < 256; u += 128) {
            const int c = u >> 2, q = u & 3;
            const uint4* s4 = (const uint4*)(wsel + (size_t)c * KT_ +
                                             (size_t)s * KSLAB + q * 32);
            uint4* d4 = (uint4*)(buf + c * COLSLOT + q * 80);
            d4[0] = s4[0]; d4[1] = s4[1]; d4[2] = s4[2]; d4[3] = s4[3];
        }
        __syncthreads();
#endif
        if (s < NXSLAB) {
            const float* ab = x + ((size_t)arow * S_ + t) * D_ + (size_t)s * KSLAB;
#pragma unroll
            for (int q = 0; q < 4; ++q) {
                v16bf a = cvt16(ab + q * 32 + khalf);
                const char* bb = buf + q * 80 + khalf * 2;
#pragma unroll
                for (int j = 0; j < 4; ++j) {
                    v16bf b = *(const v16bf*)(bb + (16 * j + ncol) * COLSLOT);
                    acc[j] = __builtin_amdgcn_wmma_f32_16x16x32_bf16(
                        false, a, false, b, (short)0, acc[j], false, false);
                }
            }
        } else {
            const __bf16* ab = rhbf + (size_t)arow * H_ + (size_t)(s - NXSLAB) * KSLAB;
#pragma unroll
            for (int q = 0; q < 4; ++q) {
                v16bf a = *(const v16bf*)(ab + q * 32 + khalf);
                const char* bb = buf + q * 80 + khalf * 2;
#pragma unroll
                for (int j = 0; j < 4; ++j) {
                    v16bf b = *(const v16bf*)(bb + (16 * j + ncol) * COLSLOT);
                    acc[j] = __builtin_amdgcn_wmma_f32_16x16x32_bf16(
                        false, a, false, b, (short)0, acc[j], false, false);
                }
            }
        }
        __syncthreads();
    }

#pragma unroll
    for (int j = 0; j < 4; ++j) {
        const int ng = n0 + 16 * j + ncol;
        const float bv = bt[ng];
#pragma unroll
        for (int i = 0; i < 8; ++i) {
            const int m = m0 + i + 8 * (lane >> 4);
            const float htil = tanhf(acc[j][i] + bv);
            const float zv   = zbuf[(size_t)m * H_ + ng];
            const float hold = hprev[(size_t)m * H_ + ng];
            const float hnew = zv * hold + (1.0f - zv) * htil;
            hnext[(size_t)m * H_ + ng]   = hnew;
            hbfnext[(size_t)m * H_ + ng] = (__bf16)hnew;
        }
    }
}

// ---- final: out[b] = sigmoid(h[b,:] . Wo + bo); one wave per batch row ----
__global__ __launch_bounds__(32) void out_kernel(const float* __restrict__ h,
                                                 const float* __restrict__ Wo,
                                                 const float* __restrict__ bo,
                                                 float* __restrict__ out) {
    const int b = blockIdx.x;
    const int lane = threadIdx.x;
    float s = 0.0f;
    for (int k = lane; k < H_; k += 32)
        s += h[(size_t)b * H_ + k] * Wo[k];
#pragma unroll
    for (int off = 16; off > 0; off >>= 1)
        s += __shfl_down(s, off, 32);
    if (lane == 0) out[b] = sigmoidf_(s + bo[0]);
}

__global__ __launch_bounds__(256) void copyh_kernel(const float* __restrict__ h,
                                                    float* __restrict__ out) {
    const int i = blockIdx.x * 256 + threadIdx.x;
    out[i] = h[i];
}

extern "C" void kernel_launch(void* const* d_in, const int* in_sizes, int n_in,
                              void* d_out, int out_size, void* d_ws, size_t ws_size,
                              hipStream_t stream) {
    (void)in_sizes; (void)n_in; (void)out_size; (void)ws_size;
    const float* x  = (const float*)d_in[0];
    const float* Wr = (const float*)d_in[1];
    const float* br = (const float*)d_in[2];
    const float* Wz = (const float*)d_in[3];
    const float* bz = (const float*)d_in[4];
    const float* Wt = (const float*)d_in[5];
    const float* bt = (const float*)d_in[6];
    const float* Wo = (const float*)d_in[7];
    const float* bo = (const float*)d_in[8];

    char* ws = (char*)d_ws;
    __bf16* WrT = (__bf16*)ws;                               // [H][KT] bf16
    __bf16* WzT = WrT + (size_t)H_ * KT_;
    __bf16* WtT = WzT + (size_t)H_ * KT_;
    float* h0 = (float*)(ws + 3 * (size_t)H_ * KT_ * sizeof(__bf16));
    float* h1 = h0 + (size_t)B_ * H_;
    float* zb = h1 + (size_t)B_ * H_;
    __bf16* hbf0 = (__bf16*)(zb + (size_t)B_ * H_);
    __bf16* hbf1 = hbf0 + (size_t)B_ * H_;
    __bf16* rhbf = hbf1 + (size_t)B_ * H_;

    wconv_kernel<<<dim3(KT_ / 16, H_ / 16, 3), dim3(16, 16), 0, stream>>>(Wr, Wz, Wt, WrT);
    zero_kernel<<<(B_ * H_) / 256, 256, 0, stream>>>(h0, hbf0);

    for (int t = 0; t < S_; ++t) {
        float*  hp  = (t & 1) ? h1 : h0;
        float*  hn  = (t & 1) ? h0 : h1;
        __bf16* hbp = (t & 1) ? hbf1 : hbf0;
        __bf16* hbn = (t & 1) ? hbf0 : hbf1;
        rz_gemm_kernel<<<dim3(4, 32), 128, 0, stream>>>(x, t, hp, hbp, WrT, WzT,
                                                        br, bz, rhbf, zb);
        ht_gemm_kernel<<<dim3(4, 16), 128, 0, stream>>>(x, t, hp, rhbf, zb,
                                                        WtT, bt, hn, hbn);
    }
    // S_ = 256 steps: last write (t=255, odd) lands in h0.
    float* hf = h0;
    float* out = (float*)d_out;
    out_kernel<<<B_, 32, 0, stream>>>(hf, Wo, bo, out);
    copyh_kernel<<<(B_ * H_) / 256, 256, 0, stream>>>(hf, out + B_);
}
